// SphGen_13314398618390
// MI455X (gfx1250) — matrix operand order repaired and checked by reading, not compiled, not run on hardware
//
#include <hip/hip_runtime.h>
#include <hip/hip_bf16.h>

// ---------------------------------------------------------------------------
// SphGen forward on gfx1250.
//  * Heavy GEMMs via v_wmma_f32_16x16x32_f16 (f16 A/B, f32 accumulate).
//  * f16 row-major A + f16 transposed weights [N][K]: each lane's WMMA
//    fragment = two contiguous global_load_b128 (ISA 7.12.2 16-bit layouts).
//  * Big-GEMM kernel: 1 wave -> 16x64 strip (4 accumulators), A fragment
//    loaded once per k-step and reused across 4 WMMAs; 4 waves/block cover
//    4 distinct M tiles => no redundant A traffic inside a block.
//  * Batch-masked attention reduced exactly to the 32 same-graph keys.
// ---------------------------------------------------------------------------

typedef _Float16 h8     __attribute__((ext_vector_type(8)));
typedef _Float16 v16h_t __attribute__((ext_vector_type(16)));
typedef float    v8f_t  __attribute__((ext_vector_type(8)));

#define ACT_NONE 0
#define ACT_RELU 1
#define ACT_TANH 2

__device__ __forceinline__ v16h_t pack16(h8 lo, h8 hi)
{
    v16h_t r;
#pragma unroll
    for (int i = 0; i < 8; ++i) { r[i] = lo[i]; r[i + 8] = hi[i]; }
    return r;
}

// ------------------------- big WMMA GEMM -------------------------------
// For M%64==0 (M=16384 paths). Block = 128 threads (4 waves).
// Wave w computes the 16x64 strip at (blockIdx.x*64 + w*16, blockIdx.y*64):
// 4 accumulators, A fragment loaded once per k-step, reused by 4 WMMAs.
__global__ void gemm_wmma_big(const _Float16* __restrict__ A, int lda,
                              const _Float16* __restrict__ Bt,   // [N][K]
                              const float* __restrict__ bias,
                              float* __restrict__ C,
                              int M, int K, int N, int act)
{
    const int wave = threadIdx.x >> 5;
    const int lane = threadIdx.x & 31;
    const int m0   = blockIdx.x * 64 + wave * 16;
    const int n0   = blockIdx.y * 64;
    const int g    = lane >> 4;
    const int rc   = lane & 15;

    const _Float16* arow = A  + (size_t)(m0 + rc) * lda;
    const _Float16* brow = Bt + (size_t)(n0 + rc) * K;   // subtile j: +16*j*K

    v8f_t acc[4] = {};

    for (int k0 = 0; k0 < K; k0 += 32) {
        // A lane fragment: K = k0 + {g*8..+7} and {16+g*8..+7}
        h8 alo = *(const h8*)(arow + k0 + g * 8);
        h8 ahi = *(const h8*)(arow + k0 + 16 + g * 8);
        v16h_t a = pack16(alo, ahi);
#pragma unroll
        for (int j = 0; j < 4; ++j) {
            const _Float16* bp = brow + (size_t)(16 * j) * K + k0 + g * 8;
            v16h_t b = pack16(*(const h8*)bp, *(const h8*)(bp + 16));
            acc[j] = __builtin_amdgcn_wmma_f32_16x16x32_f16(false, a, false, b,
                                                            (short)0, acc[j],
                                                            false, false);
        }
    }

#pragma unroll
    for (int j = 0; j < 4; ++j) {
        const int col = n0 + 16 * j + rc;
        const float bb = bias ? bias[col] : 0.f;
#pragma unroll
        for (int r = 0; r < 8; ++r) {
            int row = m0 + r + 8 * g;                 // C VGPR r: M = r + 8*g
            float v = acc[j][r] + bb;
            if (act == ACT_RELU)      v = fmaxf(v, 0.f);
            else if (act == ACT_TANH) v = tanhf(v);
            C[(size_t)row * N + col] = v;
        }
    }
}

// ------------------------- small WMMA GEMM -----------------------------
// For M=512 paths (Q/O projections, flow layer-1): wave per 16x16 tile to
// keep wave count up. Block = 128 threads (4 waves), grid (M/16, N/64).
__global__ void gemm_wmma_small(const _Float16* __restrict__ A, int lda,
                                const _Float16* __restrict__ Bt,
                                const float* __restrict__ bias,
                                float* __restrict__ C,
                                int M, int K, int N, int act)
{
    const int wave = threadIdx.x >> 5;
    const int lane = threadIdx.x & 31;
    const int m0   = blockIdx.x * 16;
    const int n0   = blockIdx.y * 64 + wave * 16;
    const int g    = lane >> 4;
    const int rc   = lane & 15;
    if (n0 >= N) return;   // wave-uniform

    const _Float16* arow = A  + (size_t)(m0 + rc) * lda;
    const _Float16* brow = Bt + (size_t)(n0 + rc) * K;

    v8f_t c = {0.f, 0.f, 0.f, 0.f, 0.f, 0.f, 0.f, 0.f};
    for (int k0 = 0; k0 < K; k0 += 32) {
        v16h_t a = pack16(*(const h8*)(arow + k0 + g * 8),
                          *(const h8*)(arow + k0 + 16 + g * 8));
        v16h_t b = pack16(*(const h8*)(brow + k0 + g * 8),
                          *(const h8*)(brow + k0 + 16 + g * 8));
        c = __builtin_amdgcn_wmma_f32_16x16x32_f16(false, a, false, b,
                                                   (short)0, c, false, false);
    }

    const float bb = bias ? bias[n0 + rc] : 0.f;
#pragma unroll
    for (int r = 0; r < 8; ++r) {
        int row = m0 + r + 8 * g;
        float v = c[r] + bb;
        if (act == ACT_RELU)      v = fmaxf(v, 0.f);
        else if (act == ACT_TANH) v = tanhf(v);
        C[(size_t)row * N + n0 + rc] = v;
    }
}

// ------------------------- prep / convert ------------------------------
__global__ void prep_node(const float* __restrict__ nf,
                          const float* __restrict__ emb,
                          const int* __restrict__ ntype,
                          _Float16* __restrict__ A16,
                          _Float16* __restrict__ E16,
                          float* __restrict__ E32)
{
    int gid = blockIdx.x * blockDim.x + threadIdx.x;
    if (gid >= 16384 * 128) return;
    int i = gid >> 7, h = gid & 127;
    float v = nf[gid];
    A16[gid] = (_Float16)v;
    int graph = i >> 5;                           // batch = i/32
    float ev = v * emb[ntype[graph] * 128 + h];
    E32[gid] = ev;
    E16[gid] = (_Float16)ev;
}

// W[K][N] f32 -> Wt[N][K] f16
__global__ void transpose_w(const float* __restrict__ W,
                            _Float16* __restrict__ Wt, int K, int N)
{
    int gid = blockIdx.x * blockDim.x + threadIdx.x;
    if (gid >= K * N) return;
    int k = gid / N, n = gid - k * N;
    Wt[(size_t)n * K + k] = (_Float16)W[gid];
}

// dst[i][doff+h] = src[idx[i*istride+ioff]][h]   (512 rows x 128 f16)
__global__ void gather16(const _Float16* __restrict__ src,
                         const int* __restrict__ idx, int istride, int ioff,
                         _Float16* __restrict__ dst, int dstride, int doff)
{
    int gid = blockIdx.x * blockDim.x + threadIdx.x;
    if (gid >= 512 * 128) return;
    int i = gid >> 7, h = gid & 127;
    int row = idx[i * istride + ioff];
    dst[(size_t)i * dstride + doff + h] = src[(size_t)row * 128 + h];
}

// feat[i][doff+h] = (f16) glob[i][h]
__global__ void copy_glob(const float* __restrict__ glob,
                          _Float16* __restrict__ dst, int dstride, int doff)
{
    int gid = blockIdx.x * blockDim.x + threadIdx.x;
    if (gid >= 512 * 128) return;
    int i = gid >> 7, h = gid & 127;
    dst[(size_t)i * dstride + doff + h] = (_Float16)glob[gid];
}

// ------------------------- attention (32 local keys, exact) ------------
__global__ void attn_local(const float* __restrict__ Q,
                           const float* __restrict__ Kp,
                           const float* __restrict__ Vp,
                           const int* __restrict__ idx, int istride, int ioff,
                           _Float16* __restrict__ preO)
{
    __shared__ float sc[128];
    const int i = blockIdx.x;
    const int t = threadIdx.x;
    const int qb = idx[i * istride + ioff] >> 5;  // batch of source node
    const int h = t >> 5, k = t & 31;

    const float* qr = Q  + (size_t)i * 128 + h * 32;
    const float* kr = Kp + (size_t)(qb * 32 + k) * 128 + h * 32;
    float s = 0.f;
#pragma unroll
    for (int d = 0; d < 32; ++d) s += qr[d] * kr[d];
    sc[t] = s * 0.17677669529663687f;             // 1/sqrt(32)
    __syncthreads();

    if (t < 4) {                                  // per-head softmax, 32 keys
        float m = -1e30f;
        for (int kk = 0; kk < 32; ++kk) m = fmaxf(m, sc[t * 32 + kk]);
        float sum = 0.f;
        for (int kk = 0; kk < 32; ++kk) {
            float e = __expf(sc[t * 32 + kk] - m);
            sc[t * 32 + kk] = e; sum += e;
        }
        float inv = 1.f / sum;
        for (int kk = 0; kk < 32; ++kk) sc[t * 32 + kk] *= inv;
    }
    __syncthreads();

    float acc = 0.f;
    const int hh = t >> 5;
#pragma unroll
    for (int kk = 0; kk < 32; ++kk)
        acc += sc[hh * 32 + kk] * Vp[(size_t)(qb * 32 + kk) * 128 + t];
    preO[(size_t)i * 128 + t] = (_Float16)acc;
}

// ------------------------- focus head ----------------------------------
__global__ void focus_l2(const float* __restrict__ hidden,   // [16384][256]
                         const float* __restrict__ w2,       // [256]
                         const float* __restrict__ b2,       // [1]
                         float* __restrict__ out)            // [16384]
{
    __shared__ float red[256];
    const int row = blockIdx.x, t = threadIdx.x;
    red[t] = hidden[(size_t)row * 256 + t] * w2[t];
    __syncthreads();
    for (int s = 128; s > 0; s >>= 1) {
        if (t < s) red[t] += red[t + s];
        __syncthreads();
    }
    if (t == 0) out[row] = 1.f / (1.f + __expf(-(red[0] + b2[0])));
}

// ------------------------- flow layer tail -----------------------------
__global__ void flow_l2(const float* __restrict__ hidden,    // [512][128] tanh'd
                        const float* __restrict__ W2,        // [128][2*odim]
                        const float* __restrict__ b2,
                        float* __restrict__ x, float* __restrict__ lj, int odim)
{
    __shared__ float st[32];
    const int i = blockIdx.x, j = threadIdx.x;
    const int o2 = 2 * odim;
    if (j < o2) {
        float acc = b2[j];
        const float* hr = hidden + (size_t)i * 128;
        for (int h = 0; h < 128; ++h) acc += hr[h] * W2[h * o2 + j];
        st[j] = acc;
    }
    __syncthreads();
    if (j < odim) {
        float s = tanhf(st[j]);
        float t = st[j + odim];
        size_t p = (size_t)i * odim + j;
        x[p]  = (x[p] + t) * __expf(s);
        lj[p] += s;
    }
}

// ------------------------- flow state init -----------------------------
__global__ void init_node_x(const int* __restrict__ ntype,
                            const float* __restrict__ noise,
                            float* __restrict__ x, float* __restrict__ lj)
{
    int gid = blockIdx.x * blockDim.x + threadIdx.x;
    if (gid >= 512 * 10) return;
    int i = gid / 10, j = gid - i * 10;
    x[gid] = (ntype[i] == j ? 1.f : 0.f) + 0.9f * noise[gid];
    lj[gid] = 0.f;
}

__global__ void init_scalar_x(const float* nd, const float* na, const float* nt,
                              float* dx, float* dlj, float* ax, float* alj,
                              float* tx, float* tlj)
{
    int i = blockIdx.x * blockDim.x + threadIdx.x;
    if (i >= 512) return;
    dx[i] = nd[i]; dlj[i] = 0.f;
    ax[i] = na[i]; alj[i] = 0.f;
    tx[i] = nt[i]; tlj[i] = 0.f;
}

// ===========================================================================
extern "C" void kernel_launch(void* const* d_in, const int* in_sizes, int n_in,
                              void* d_out, int out_size, void* d_ws, size_t ws_size,
                              hipStream_t stream)
{
    (void)in_sizes; (void)n_in; (void)out_size;

    // ---- inputs (setup_inputs dict order; params depth-first insertion order)
    const float* node_feat = (const float*)d_in[0];
    const float* emb_table = (const float*)d_in[1];
    const int*   focus     = (const int*)d_in[3];   // [512][2]
    const int*   c1f       = (const int*)d_in[4];   // [512][2]
    const int*   c2f       = (const int*)d_in[5];   // [512][3]
    const int*   ntype     = (const int*)d_in[6];
    const float* new_dist  = (const float*)d_in[7];
    const float* new_angle = (const float*)d_in[8];
    const float* new_tor   = (const float*)d_in[9];
    const float* noise     = (const float*)d_in[10];

    const int P = 11;
    const float* f_b1 = (const float*)d_in[P + 1];
    const float* f_w2 = (const float*)d_in[P + 2];  // [256][1]
    const float* f_b2 = (const float*)d_in[P + 3];
    // atts: node=15, dist=23, angle=31, tor=39; within: qw,qb,kw,kb,vw,vb,ow,ob
    const int attBase[4] = {P + 4, P + 12, P + 20, P + 28};
    // flows: node=47, dist=71, angle=95, tor=119; per layer: l1w,l1b,l2w,l2b
    const int flowBase[4] = {P + 36, P + 60, P + 84, P + 108};

    // ---- outputs
    float* out = (float*)d_out;
    float* out_node_x  = out;            // 512*10
    float* out_node_lj = out + 5120;     // 512*10
    float* out_focus   = out + 10240;    // 16384
    float* out_dx      = out + 26624;    float* out_dlj = out + 27136;
    float* out_ax      = out + 27648;    float* out_alj = out + 28160;
    float* out_tx      = out + 28672;    float* out_tlj = out + 29184;

    // ---- workspace bump allocator
    char* wp = (char*)d_ws;
    auto alloc = [&](size_t bytes) -> void* {
        void* p = (void*)wp;
        wp += (bytes + 255) & ~(size_t)255;
        return p;
    };
    _Float16* A16   = (_Float16*)alloc(16384 * 128 * 2);   // node_feat f16
    _Float16* E16   = (_Float16*)alloc(16384 * 128 * 2);   // node_emb f16
    float*    E32   = (float*)   alloc((size_t)16384 * 128 * 4);
    float*    Hfoc  = (float*)   alloc((size_t)16384 * 256 * 4);
    float*    Kp    = (float*)   alloc((size_t)16384 * 128 * 4);
    float*    Vp    = (float*)   alloc((size_t)16384 * 128 * 4);
    float*    Qp    = (float*)   alloc(512 * 128 * 4);
    _Float16* preO  = (_Float16*)alloc(512 * 128 * 2);
    float*    glob  = (float*)   alloc(512 * 128 * 4);
    float*    Hflow = (float*)   alloc(512 * 128 * 4);

    const int fK[4]   = {256, 256, 384, 512};  // feat dims (node,dist,angle,tor)
    const int qKd[4]  = {128, 128, 256, 384};  // q-proj input dims
    const int fOD[4]  = {10, 1, 1, 1};
    _Float16* feat[4];
    for (int a = 0; a < 4; ++a) feat[a] = (_Float16*)alloc((size_t)512 * fK[a] * 2);

    _Float16* WTf1 = (_Float16*)alloc(256 * 128 * 2);      // focus l1^T
    _Float16 *WTq[4], *WTk[4], *WTv[4], *WTo[4];
    for (int a = 0; a < 4; ++a) {
        WTq[a] = (_Float16*)alloc((size_t)128 * qKd[a] * 2);
        WTk[a] = (_Float16*)alloc(128 * 128 * 2);
        WTv[a] = (_Float16*)alloc(128 * 128 * 2);
        WTo[a] = (_Float16*)alloc(128 * 128 * 2);
    }
    _Float16* WTfl[4][6];
    for (int f = 0; f < 4; ++f)
        for (int l = 0; l < 6; ++l)
            WTfl[f][l] = (_Float16*)alloc((size_t)128 * fK[f] * 2);

    if ((size_t)(wp - (char*)d_ws) > ws_size) return;  // ws too small: bail

    // ---- helpers
    auto xpose = [&](const void* W, _Float16* Wt, int K, int N) {
        int tot = K * N;
        transpose_w<<<(tot + 255) / 256, 256, 0, stream>>>((const float*)W, Wt, K, N);
    };
    auto gemmBig = [&](const _Float16* A, int lda, const _Float16* Bt,
                       const float* bias, float* C, int M, int K, int N, int act) {
        gemm_wmma_big<<<dim3(M / 64, N / 64), 128, 0, stream>>>(A, lda, Bt, bias, C, M, K, N, act);
    };
    auto gemmSmall = [&](const _Float16* A, int lda, const _Float16* Bt,
                         const float* bias, float* C, int M, int K, int N, int act) {
        gemm_wmma_small<<<dim3(M / 16, N / 64), 128, 0, stream>>>(A, lda, Bt, bias, C, M, K, N, act);
    };

    // ---- stage 0: conversions
    prep_node<<<(16384 * 128 + 255) / 256, 256, 0, stream>>>(
        node_feat, emb_table, ntype, A16, E16, E32);

    xpose(d_in[P + 0], WTf1, 128, 256);
    for (int a = 0; a < 4; ++a) {
        int b = attBase[a];
        xpose(d_in[b + 0], WTq[a], qKd[a], 128);
        xpose(d_in[b + 2], WTk[a], 128, 128);
        xpose(d_in[b + 4], WTv[a], 128, 128);
        xpose(d_in[b + 6], WTo[a], 128, 128);
    }
    for (int f = 0; f < 4; ++f)
        for (int l = 0; l < 6; ++l)
            xpose(d_in[flowBase[f] + 4 * l], WTfl[f][l], fK[f], 128);

    // ---- flow state init (lives directly in d_out)
    init_node_x<<<(5120 + 255) / 256, 256, 0, stream>>>(ntype, noise, out_node_x, out_node_lj);
    init_scalar_x<<<2, 256, 0, stream>>>(new_dist, new_angle, new_tor,
                                         out_dx, out_dlj, out_ax, out_alj, out_tx, out_tlj);

    // ---- focus MLP: relu(node_feat@W1+b1) -> sigmoid(.@w2+b2)
    gemmBig(A16, 128, WTf1, f_b1, Hfoc, 16384, 128, 256, ACT_RELU);
    focus_l2<<<16384, 256, 0, stream>>>(Hfoc, f_w2, f_b2, out_focus);

    // ---- build local feature halves (f16, also serve as q-proj inputs)
    gather16<<<256, 256, 0, stream>>>(A16, focus, 2, 0, feat[0], 256, 0);   // local_node
    gather16<<<256, 256, 0, stream>>>(E16, focus, 2, 0, feat[1], 256, 0);   // local_dist
    gather16<<<256, 256, 0, stream>>>(E16, c1f, 2, 1, feat[2], 384, 0);     // local_angle
    gather16<<<256, 256, 0, stream>>>(E16, c1f, 2, 0, feat[2], 384, 128);
    gather16<<<256, 256, 0, stream>>>(E16, c2f, 3, 2, feat[3], 512, 0);     // local_tor
    gather16<<<256, 256, 0, stream>>>(E16, c2f, 3, 1, feat[3], 512, 128);
    gather16<<<256, 256, 0, stream>>>(E16, c2f, 3, 0, feat[3], 512, 256);

    // ---- four attentions
    const _Float16* kvSrc[4] = {A16, E16, E16, E16};
    const int* qIdx[4]    = {focus, focus, c1f, c2f};
    const int  qStride[4] = {2, 2, 2, 3};

    for (int a = 0; a < 4; ++a) {
        int b = attBase[a];
        const float* qb_ = (const float*)d_in[b + 1];
        const float* kb_ = (const float*)d_in[b + 3];
        const float* vb_ = (const float*)d_in[b + 5];
        const float* ob_ = (const float*)d_in[b + 7];

        gemmBig(kvSrc[a], 128, WTk[a], kb_, Kp, 16384, 128, 128, ACT_NONE);
        gemmBig(kvSrc[a], 128, WTv[a], vb_, Vp, 16384, 128, 128, ACT_NONE);
        gemmSmall(feat[a], fK[a], WTq[a], qb_, Qp, 512, qKd[a], 128, ACT_NONE);
        attn_local<<<512, 128, 0, stream>>>(Qp, Kp, Vp, qIdx[a], qStride[a], 0, preO);
        gemmSmall(preO, 128, WTo[a], ob_, glob, 512, 128, 128, ACT_NONE);
        copy_glob<<<256, 256, 0, stream>>>(glob, feat[a], fK[a], fK[a] - 128);
    }

    // ---- four affine-coupling flows (6 layers each)
    float* fx[4]  = {out_node_x,  out_dx,  out_ax,  out_tx};
    float* flj[4] = {out_node_lj, out_dlj, out_alj, out_tlj};
    for (int f = 0; f < 4; ++f) {
        for (int l = 0; l < 6; ++l) {
            int b = flowBase[f] + 4 * l;
            const float* l1b = (const float*)d_in[b + 1];
            const float* l2w = (const float*)d_in[b + 2];
            const float* l2b = (const float*)d_in[b + 3];
            gemmSmall(feat[f], fK[f], WTfl[f][l], l1b, Hflow, 512, fK[f], 128, ACT_TANH);
            flow_l2<<<512, 32, 0, stream>>>(Hflow, l2w, l2b, fx[f], flj[f], fOD[f]);
        }
    }
}
